// Multi_Head_Attention_86088324481479
// MI455X (gfx1250) — matrix operand order repaired and checked by reading, not compiled, not run on hardware
//
#include <hip/hip_runtime.h>
#include <hip/hip_bf16.h>

// ---------------------------------------------------------------------------
// MHA for MI455X (gfx1250, wave32). All matmuls via v_wmma_f32_16x16x32_bf16.
// - One-shot fp32->bf16 conversion of activations + weights (bandwidth-cheap,
//   removes redundant conversion VALU from the GEMM hot loops).
// - 6 GEMM+bias launches: 128x128 block tile, 8 waves, wave tile 32x64,
//   packed-dword register-staged pipeline + global_prefetch of tile k+2.
// - Flash attention: 64-key tiles (DPP butterfly cost amortized 2x), online
//   softmax with DPP16 reductions, pipelined K/V staging, 16 WMMA/iter/wave.
// ---------------------------------------------------------------------------

#define D_MODEL 1024
#define SEQ     2048
#define BATCH   2
#define HEADS   16
#define KDIM    64

typedef __attribute__((ext_vector_type(16))) __bf16 v16bf;
typedef __attribute__((ext_vector_type(8)))  float  v8f;

union BFrag {
    v16bf v;
    unsigned u[8];
};

__device__ __forceinline__ unsigned short f2bf(float f) {
    unsigned u = __float_as_uint(f);
    unsigned r = u + 0x7FFFu + ((u >> 16) & 1u);   // round-to-nearest-even
    return (unsigned short)(r >> 16);
}

// ---- DPP16 butterfly reduction within each 16-lane row (wave32) ------------
template <int CTRL>
__device__ __forceinline__ float dppmov(float x) {
    return __uint_as_float((unsigned)__builtin_amdgcn_update_dpp(
        0, (int)__float_as_uint(x), CTRL, 0xF, 0xF, true));
}
__device__ __forceinline__ float rowmax16(float x) {
    x = fmaxf(x, dppmov<0xB1>(x));    // quad_perm(1,0,3,2)
    x = fmaxf(x, dppmov<0x4E>(x));    // quad_perm(2,3,0,1)
    x = fmaxf(x, dppmov<0x141>(x));   // row_half_mirror
    x = fmaxf(x, dppmov<0x140>(x));   // row_mirror
    return x;
}
__device__ __forceinline__ float rowsum16(float x) {
    x += dppmov<0xB1>(x);
    x += dppmov<0x4E>(x);
    x += dppmov<0x141>(x);
    x += dppmov<0x140>(x);
    return x;
}

// ---------------------------------------------------------------------------
// fp32 -> bf16 packed conversion (2 elements per thread)
// ---------------------------------------------------------------------------
__global__ __launch_bounds__(256)
void cvt_f32_bf16(const float* __restrict__ in, unsigned* __restrict__ out,
                  int npairs)
{
    int i = blockIdx.x * blockDim.x + threadIdx.x;
    if (i < npairs) {
        float2 v = ((const float2*)in)[i];
        out[i] = (unsigned)f2bf(v.x) | ((unsigned)f2bf(v.y) << 16);
    }
}

// ---------------------------------------------------------------------------
// GEMM: C[M,N](bf16) = A[M,K](bf16) @ W[K,N](bf16) + bias[N](f32)
// Block tile 128x128, 8 waves (4m x 2n), wave tile 32x64 (2x4 WMMA), K-step 32.
// Register-staged pipeline + cache prefetch of the k+2 tile.
// ---------------------------------------------------------------------------
__global__ __launch_bounds__(256)
void gemm_bias_bf16(const unsigned short* __restrict__ A,
                    const unsigned short* __restrict__ Wb,
                    const float* __restrict__ bias,
                    unsigned short* __restrict__ C,
                    int M, int K, int N)
{
    __shared__ __align__(16) unsigned short As[128 * 34];    // [m][k]
    __shared__ __align__(16) unsigned short Bst[128 * 34];   // [n][k] (W^T)

    const int t    = threadIdx.x;
    const int lane = t & 31;
    const int wv   = t >> 5;
    const int wm   = wv >> 1;          // 0..3 (rows, 32 each)
    const int wn   = wv & 1;           // 0..1 (cols, 64 each)
    const int m0   = blockIdx.y * 128;
    const int n0   = blockIdx.x * 128;
    const int l16  = lane & 15;
    const int half = lane >> 4;

    unsigned aS[8], bS[8];             // packed bf16 pairs

    auto loadA = [&](int k0) {
        #pragma unroll
        for (int i = 0; i < 8; ++i) {
            int e = t + i * 256, r = e >> 4, cw = e & 15;   // 128 x 16 dwords
            aS[i] = *(const unsigned*)(A + (size_t)(m0 + r) * K + k0 + 2 * cw);
        }
    };
    auto loadB = [&](int k0) {
        #pragma unroll
        for (int i = 0; i < 8; ++i) {
            int e = t + i * 256, kr = e >> 6, ncw = e & 63; // 32 x 64 dwords
            bS[i] = *(const unsigned*)(Wb + (size_t)(k0 + kr) * N + n0 + 2 * ncw);
        }
    };

    v8f acc[2][4] = {};

    loadA(0);
    loadB(0);

    for (int k0 = 0; k0 < K; k0 += 32) {
        __syncthreads();
        // Commit staged registers to LDS
        #pragma unroll
        for (int i = 0; i < 8; ++i) {
            int e = t + i * 256, r = e >> 4, cw = e & 15;
            *(unsigned*)(As + r * 34 + 2 * cw) = aS[i];
        }
        #pragma unroll
        for (int i = 0; i < 8; ++i) {
            int e = t + i * 256, kr = e >> 6, ncw = e & 63;
            Bst[(2 * ncw) * 34 + kr]     = (unsigned short)(bS[i] & 0xFFFFu);
            Bst[(2 * ncw + 1) * 34 + kr] = (unsigned short)(bS[i] >> 16);
        }
        __syncthreads();
        // Pipeline: next tile into registers; warm L2->WGP for tile k+2
        if (k0 + 32 < K) {
            loadA(k0 + 32);
            loadB(k0 + 32);
            if (k0 + 64 < K) {
                int r = t >> 4, cw = t & 15;
                __builtin_prefetch(A + (size_t)(m0 + r) * K + k0 + 64 + 2 * cw, 0, 3);
                int kr = t >> 6, ncw = t & 63;
                __builtin_prefetch(Wb + (size_t)(k0 + 64 + kr) * N + n0 + 2 * ncw, 0, 3);
            }
        }

        // A fragments (2 m-tiles), live across the nt loop
        BFrag a[2];
        #pragma unroll
        for (int mt = 0; mt < 2; ++mt) {
            const unsigned short* base = &As[(wm * 32 + mt * 16 + l16) * 34];
            #pragma unroll
            for (int vv = 0; vv < 8; ++vv) {
                int kk = (vv < 4) ? (half * 8 + 2 * vv)
                                  : (16 + half * 8 + 2 * (vv - 4));
                a[mt].u[vv] = *(const unsigned*)(base + kk);
            }
        }
        #pragma unroll
        for (int nt = 0; nt < 4; ++nt) {
            BFrag b;
            const unsigned short* base = &Bst[(wn * 64 + nt * 16 + l16) * 34];
            #pragma unroll
            for (int vv = 0; vv < 8; ++vv)
                b.u[vv] = *(const unsigned*)(base + ((vv < 4) ? (half * 8 + 2 * vv) : (16 + half * 8 + 2 * (vv - 4))));
            #pragma unroll
            for (int mt = 0; mt < 2; ++mt)
                acc[mt][nt] = __builtin_amdgcn_wmma_f32_16x16x32_bf16(
                    false, a[mt].v, false, b.v, (short)0, acc[mt][nt],
                    false, false);
        }
    }

    // Epilogue: +bias, store bf16
    #pragma unroll
    for (int nt = 0; nt < 4; ++nt) {
        int   col = n0 + wn * 64 + nt * 16 + l16;
        float bv  = bias[col];
        #pragma unroll
        for (int mt = 0; mt < 2; ++mt)
            #pragma unroll
            for (int r = 0; r < 8; ++r) {
                int row = m0 + wm * 32 + mt * 16 + half * 8 + r;
                C[(size_t)row * N + col] = f2bf(acc[mt][nt][r] + bv);
            }
    }
}

// ---------------------------------------------------------------------------
// Flash attention: one block per (b, h, 64-row Q tile); 4 waves x 16 Q rows.
// 64 keys per iteration: 8 QK^T WMMAs + 8 P@V WMMAs per wave per iteration.
// ---------------------------------------------------------------------------
__global__ __launch_bounds__(128)
void flash_attn_bf16(const unsigned short* __restrict__ Q,
                     const unsigned short* __restrict__ Km,
                     const unsigned short* __restrict__ Vm,
                     float* __restrict__ Out)
{
    __shared__ __align__(16) unsigned short Ks[64 * 66];       // [key][kd]
    __shared__ __align__(16) unsigned short Vs[64 * 66];       // [kd][key] (T)
    __shared__ __align__(16) unsigned short Ps[4 * 16 * 66];   // per-wave P

    const int bid  = blockIdx.x;
    const int tile = bid & 31;               // SEQ/64 = 32
    const int h    = (bid >> 5) & 15;
    const int b    = bid >> 9;
    const int t    = threadIdx.x;
    const int lane = t & 31;
    const int wv   = t >> 5;
    const int l16  = lane & 15;
    const int half = lane >> 4;
    const int qbase = tile * 64 + wv * 16;

    // Q fragments (Kd=64 -> 2 frags), live in registers for the whole loop
    BFrag qf[2];
    {
        const size_t rowoff =
            ((size_t)(b * SEQ + qbase + l16)) * D_MODEL + h * KDIM;
        #pragma unroll
        for (int f = 0; f < 2; ++f)
            #pragma unroll
            for (int vv = 0; vv < 8; ++vv) {
                int kk = f * 32 + ((vv < 4) ? (half * 8 + 2 * vv)
                                            : (16 + half * 8 + 2 * (vv - 4)));
                qf[f].u[vv] = *(const unsigned*)(Q + rowoff + kk);
            }
    }

    v8f   o[4] = {};
    float mrow[8], lrow[8];
    #pragma unroll
    for (int r = 0; r < 8; ++r) { mrow[r] = -1e30f; lrow[r] = 0.0f; }

    unsigned short* Pw = &Ps[wv * 16 * 66];

    // Register-staged K/V tile pipeline: 64 keys x 64 kd = 2048 dwords
    unsigned kReg[16], vReg[16];
    auto loadKV = [&](int kt) {
        #pragma unroll
        for (int i = 0; i < 16; ++i) {
            int e = t + i * 128;
            int key = e >> 5, kdw = e & 31;
            size_t off = ((size_t)(b * SEQ + kt + key)) * D_MODEL + h * KDIM + 2 * kdw;
            kReg[i] = *(const unsigned*)(Km + off);
            vReg[i] = *(const unsigned*)(Vm + off);
        }
    };
    loadKV(0);

    for (int kt = 0; kt < SEQ; kt += 64) {
        __syncthreads();
        // Commit staged K tile [64 keys][64 kd] and V tile transposed [kd][key]
        #pragma unroll
        for (int i = 0; i < 16; ++i) {
            int e = t + i * 128;
            int key = e >> 5, kdw = e & 31;
            *(unsigned*)(Ks + key * 66 + 2 * kdw) = kReg[i];
            Vs[(2 * kdw) * 66 + key]     = (unsigned short)(vReg[i] & 0xFFFFu);
            Vs[(2 * kdw + 1) * 66 + key] = (unsigned short)(vReg[i] >> 16);
        }
        __syncthreads();
        if (kt + 64 < SEQ) loadKV(kt + 64);        // overlap with math below

        // Scores: four 16-key n-tiles, each 2 WMMAs over Kd=64
        v8f sc[4];
        #pragma unroll
        for (int j = 0; j < 4; ++j) {
            v8f c = {};
            const unsigned short* kb = &Ks[(j * 16 + l16) * 66];
            #pragma unroll
            for (int f = 0; f < 2; ++f) {
                BFrag bf;
                #pragma unroll
                for (int vv = 0; vv < 8; ++vv)
                    bf.u[vv] = *(const unsigned*)(kb + f * 32 + ((vv < 4) ? (half * 8 + 2 * vv) : (16 + half * 8 + 2 * (vv - 4))));
                c = __builtin_amdgcn_wmma_f32_16x16x32_bf16(
                    false, qf[f].v, false, bf.v, (short)0, c, false, false);
            }
            sc[j] = c;
        }

        // Online softmax: per-row reductions via DPP16 butterflies
        float scale[8];
        #pragma unroll
        for (int r = 0; r < 8; ++r) {
            float s0 = sc[0][r] * 0.125f;          // 1/sqrt(64)
            float s1 = sc[1][r] * 0.125f;
            float s2 = sc[2][r] * 0.125f;
            float s3 = sc[3][r] * 0.125f;
            float tm = rowmax16(fmaxf(fmaxf(s0, s1), fmaxf(s2, s3)));
            float newm = fmaxf(mrow[r], tm);
            scale[r]   = __expf(mrow[r] - newm);
            mrow[r]    = newm;
            float p0 = __expf(s0 - newm);
            float p1 = __expf(s1 - newm);
            float p2 = __expf(s2 - newm);
            float p3 = __expf(s3 - newm);
            sc[0][r] = p0; sc[1][r] = p1; sc[2][r] = p2; sc[3][r] = p3;
            float ps = rowsum16((p0 + p1) + (p2 + p3));
            lrow[r] = lrow[r] * scale[r] + ps;
        }
        #pragma unroll
        for (int nt = 0; nt < 4; ++nt)
            #pragma unroll
            for (int r = 0; r < 8; ++r)
                o[nt][r] *= scale[r];

        // P: C-layout -> row-major bf16 in per-wave LDS (16 x 64)
        #pragma unroll
        for (int j = 0; j < 4; ++j)
            #pragma unroll
            for (int r = 0; r < 8; ++r)
                Pw[(half * 8 + r) * 66 + j * 16 + l16] = f2bf(sc[j][r]);

        // Re-read P as two A fragments (16x32 each, keys 0-31 / 32-63)
        BFrag pf[2];
        {
            const unsigned short* pb = &Pw[l16 * 66];
            #pragma unroll
            for (int f = 0; f < 2; ++f)
                #pragma unroll
                for (int vv = 0; vv < 8; ++vv) {
                    int kk = f * 32 + ((vv < 4) ? (half * 8 + 2 * vv)
                                                : (16 + half * 8 + 2 * (vv - 4)));
                    pf[f].u[vv] = *(const unsigned*)(pb + kk);
                }
        }

        // O += P @ V  (4 n-tiles of 16 kd columns, K=64 -> 2 WMMAs each)
        #pragma unroll
        for (int nt = 0; nt < 4; ++nt) {
            const unsigned short* vb = &Vs[(nt * 16 + l16) * 66];
            #pragma unroll
            for (int f = 0; f < 2; ++f) {
                BFrag bf;
                #pragma unroll
                for (int vv = 0; vv < 8; ++vv)
                    bf.u[vv] = *(const unsigned*)(vb + f * 32 + ((vv < 4) ? (half * 8 + 2 * vv) : (16 + half * 8 + 2 * (vv - 4))));
                o[nt] = __builtin_amdgcn_wmma_f32_16x16x32_bf16(
                    false, pf[f].v, false, bf.v, (short)0, o[nt], false, false);
            }
        }
    }

    // Normalize and store f32 output [B,S,H*Kd]
    #pragma unroll
    for (int nt = 0; nt < 4; ++nt)
        #pragma unroll
        for (int r = 0; r < 8; ++r) {
            int row = qbase + half * 8 + r;
            int col = h * KDIM + nt * 16 + l16;
            Out[((size_t)(b * SEQ + row)) * D_MODEL + col] = o[nt][r] / lrow[r];
        }
}

// ---------------------------------------------------------------------------
extern "C" void kernel_launch(void* const* d_in, const int* in_sizes, int n_in,
                              void* d_out, int out_size, void* d_ws, size_t ws_size,
                              hipStream_t stream) {
    const float* x   = (const float*)d_in[0];
    const float* y   = (const float*)d_in[1];
    const float* Wq  = (const float*)d_in[2];
    const float* bq  = (const float*)d_in[3];
    const float* Wk  = (const float*)d_in[4];
    const float* bk  = (const float*)d_in[5];
    const float* Wv  = (const float*)d_in[6];
    const float* bv  = (const float*)d_in[7];
    const float* Wqm = (const float*)d_in[8];
    const float* bqm = (const float*)d_in[9];
    const float* Wkm = (const float*)d_in[10];
    const float* bkm = (const float*)d_in[11];
    const float* Wvm = (const float*)d_in[12];
    const float* bvm = (const float*)d_in[13];
    float* out = (float*)d_out;

    const int M = BATCH * SEQ;      // 4096
    const int K = D_MODEL;          // 1024
    const int N = D_MODEL;          // 1024

    // Workspace layout (bf16 / ushort elements):
    //   xb(4M) yb(4M) 6 weights(1M each) tmp q2 k2 v2 (4M each)  = 60 MB
    unsigned short* p = (unsigned short*)d_ws;
    unsigned short* xb   = p;  p += (size_t)M * K;
    unsigned short* yb   = p;  p += (size_t)M * K;
    unsigned short* wqb  = p;  p += (size_t)K * N;
    unsigned short* wkb  = p;  p += (size_t)K * N;
    unsigned short* wvb  = p;  p += (size_t)K * N;
    unsigned short* wqmb = p;  p += (size_t)K * N;
    unsigned short* wkmb = p;  p += (size_t)K * N;
    unsigned short* wvmb = p;  p += (size_t)K * N;
    unsigned short* tmp  = p;  p += (size_t)M * N;
    unsigned short* q2   = p;  p += (size_t)M * N;
    unsigned short* k2   = p;  p += (size_t)M * N;
    unsigned short* v2   = p;

    // One-shot fp32 -> bf16 conversions
    {
        const int actPairs = M * K / 2;     // 2M pairs
        const int wPairs   = K * N / 2;     // 0.5M pairs
        dim3 cb(256);
        cvt_f32_bf16<<<dim3((actPairs + 255) / 256), cb, 0, stream>>>(x,   (unsigned*)xb,   actPairs);
        cvt_f32_bf16<<<dim3((actPairs + 255) / 256), cb, 0, stream>>>(y,   (unsigned*)yb,   actPairs);
        cvt_f32_bf16<<<dim3((wPairs  + 255) / 256), cb, 0, stream>>>(Wq,  (unsigned*)wqb,  wPairs);
        cvt_f32_bf16<<<dim3((wPairs  + 255) / 256), cb, 0, stream>>>(Wk,  (unsigned*)wkb,  wPairs);
        cvt_f32_bf16<<<dim3((wPairs  + 255) / 256), cb, 0, stream>>>(Wv,  (unsigned*)wvb,  wPairs);
        cvt_f32_bf16<<<dim3((wPairs  + 255) / 256), cb, 0, stream>>>(Wqm, (unsigned*)wqmb, wPairs);
        cvt_f32_bf16<<<dim3((wPairs  + 255) / 256), cb, 0, stream>>>(Wkm, (unsigned*)wkmb, wPairs);
        cvt_f32_bf16<<<dim3((wPairs  + 255) / 256), cb, 0, stream>>>(Wvm, (unsigned*)wvmb, wPairs);
    }

    dim3 ggrid(N / 128, M / 128);   // (8, 32)
    dim3 gblk(256);

    // q2 = (x@Wq+bq) @ Wqm + bqm
    gemm_bias_bf16<<<ggrid, gblk, 0, stream>>>(xb,  wqb,  bq,  tmp, M, K, N);
    gemm_bias_bf16<<<ggrid, gblk, 0, stream>>>(tmp, wqmb, bqm, q2,  M, K, N);
    // k2 = (y@Wk+bk) @ Wkm + bkm
    gemm_bias_bf16<<<ggrid, gblk, 0, stream>>>(yb,  wkb,  bk,  tmp, M, K, N);
    gemm_bias_bf16<<<ggrid, gblk, 0, stream>>>(tmp, wkmb, bkm, k2,  M, K, N);
    // v2 = (y@Wv+bv) @ Wvm + bvm
    gemm_bias_bf16<<<ggrid, gblk, 0, stream>>>(yb,  wvb,  bv,  tmp, M, K, N);
    gemm_bias_bf16<<<ggrid, gblk, 0, stream>>>(tmp, wvmb, bvm, v2,  M, K, N);

    // Attention: B*H*(S/64) = 1024 blocks, 4 waves each
    dim3 agrid(BATCH * HEADS * (SEQ / 64));
    flash_attn_bf16<<<agrid, dim3(128), 0, stream>>>(q2, k2, v2, out);
}
